// MultiHeadLinearAttention_48919677501889
// MI455X (gfx1250) — matrix operand order repaired and checked by reading, not compiled, not run on hardware
//
#include <hip/hip_runtime.h>
#include <hip/hip_bf16.h>
#include <math.h>

// ---------------- problem constants ----------------
constexpr int Bc = 4;
constexpr int Lc = 4096;
constexpr int Cc = 1024;
constexpr int Hc = 16;
constexpr int Dc = 64;
constexpr int BLc = Bc * Lc;          // 16384 rows
constexpr float EPSc = 1e-6f;

// ---------------- types ----------------
typedef __attribute__((ext_vector_type(16))) __bf16 v16bf;
typedef __attribute__((ext_vector_type(8)))  float  v8f;
typedef __attribute__((ext_vector_type(4)))  unsigned int v4u;
typedef __attribute__((ext_vector_type(4)))  unsigned int u32x4;
typedef __attribute__((ext_vector_type(8)))  int i32x8;
typedef __attribute__((ext_vector_type(4)))  int i32x4;

#if __has_builtin(__builtin_amdgcn_tensor_load_to_lds) && __has_builtin(__builtin_amdgcn_s_wait_tensorcnt)
#define USE_TDM 1
#else
#define USE_TDM 0
#endif

// Pack two fp32 -> two bf16 in one dword
__device__ __forceinline__ unsigned pack2bf(float a, float b) {
    union { __bf16 h[2]; unsigned u; } t;
    t.h[0] = (__bf16)a; t.h[1] = (__bf16)b;
    return t.u;
}

// A fragment (16x32 bf16, MxK), row-major tile, leading dim ld (elements).
// lanes 0-15 -> M=lane, K {0-7,16-23}; lanes 16-31 -> M=lane-16, K {8-15,24-31}
__device__ __forceinline__ v16bf load_frag_a(const __bf16* base, int ld) {
    const int lane = threadIdx.x & 31;
    const int row  = lane & 15;
    const int hi   = lane >> 4;
    const __bf16* p = base + row * ld;
    union { v16bf v; v4u u[2]; } f;
    f.u[0] = *(const v4u*)(p + hi * 8);
    f.u[1] = *(const v4u*)(p + 16 + hi * 8);
    return f.v;
}

// B fragment (32x16 bf16, KxN) from K-transposed tile Bt[n][k] (row-major, ld elems).
__device__ __forceinline__ v16bf load_frag_b(const __bf16* baseT, int ld) {
    const int lane = threadIdx.x & 31;
    const int col  = lane & 15;
    const int hi   = lane >> 4;
    const __bf16* p = baseT + col * ld + hi * 8;
    union { v16bf v; v4u u[2]; } f;
    f.u[0] = *(const v4u*)(p);
    f.u[1] = *(const v4u*)(p + 16);
    return f.v;
}

__device__ __forceinline__ v8f wmma_bf16(v16bf a, v16bf b, v8f c) {
    return __builtin_amdgcn_wmma_f32_16x16x32_bf16(false, a, false, b, (short)0, c,
                                                   false, false);
}

#if USE_TDM
// Issue a TDM 2D tile load: tile_k elements per row (bf16), tile_rows rows,
// source row stride = row_stride elements. Destination = LDS byte offset lds_off.
__device__ __forceinline__ void tdm_load_2d(unsigned lds_off, const __bf16* gptr,
                                            unsigned tile_k, unsigned tile_rows,
                                            unsigned row_stride) {
    unsigned long long ga = (unsigned long long)(size_t)gptr;
    u32x4 g0;
    g0[0] = 1u;                                   // count=1 (valid descriptor)
    g0[1] = lds_off;                              // lds_addr
    g0[2] = (unsigned)(ga & 0xFFFFFFFFu);         // global_addr[31:0]
    g0[3] = (unsigned)((ga >> 32) & 0x01FFFFFFu)  // global_addr[56:32]
            | (2u << 30);                         // type=2 ("image")
    i32x8 g1 = {};
    g1[0] = 0x00010000;                           // data_size=1 (2 bytes/elem)
    const unsigned td0 = (unsigned)Cc;            // tensor_dim0 (bits 79:48)
    g1[1] = (int)((td0 & 0xFFFFu) << 16);
    g1[2] = (int)((td0 >> 16) & 0xFFFFu);
    const unsigned td1 = 1u << 20;                // tensor_dim1 (bits 111:80), large
    g1[2] |= (int)((td1 & 0xFFFFu) << 16);
    g1[3]  = (int)((td1 >> 16) & 0xFFFFu);
    g1[3] |= (int)(tile_k << 16);                 // tile_dim0 (bits 127:112)
    g1[4]  = (int)(tile_rows & 0xFFFFu);          // tile_dim1 (bits 143:128), tile_dim2=0
    const unsigned long long st = (unsigned long long)row_stride; // dim0_stride (bits 207:160)
    g1[5] = (int)(st & 0xFFFFFFFFu);
    g1[6] = (int)((st >> 32) & 0xFFFFu);
    i32x4 z4 = {};
#if defined(__clang_major__) && (__clang_major__ >= 23)
    i32x8 z8 = {};
    __builtin_amdgcn_tensor_load_to_lds(g0, g1, z4, z4, z8, 0);
#else
    __builtin_amdgcn_tensor_load_to_lds(g0, g1, z4, z4, 0);
#endif
}
__device__ __forceinline__ unsigned lds_off_of(const void* p) {
    return (unsigned)(size_t)p;                   // LDS addr = flat addr[31:0]
}
#endif

// =====================================================================
// Kernel 0: fp32 -> bf16 conversion (one shot; keeps GEMM hot loop pure bf16)
// =====================================================================
__global__ __launch_bounds__(256)
void cvt_kernel(const float* __restrict__ in, __bf16* __restrict__ out, int n4) {
    const int i = blockIdx.x * 256 + threadIdx.x;
    if (i >= n4) return;
    const float4 x = ((const float4*)in)[i];
    unsigned* o = (unsigned*)(out + (size_t)i * 4);
    o[0] = pack2bf(x.x, x.y);
    o[1] = pack2bf(x.z, x.w);
}

// =====================================================================
// Kernel 1: projection GEMM  Out[m,n] = f( X[m,:] @ W[n,:] + bias[n] )
//   X: bf16 [BL, C]  W: bf16 [C, C] row-major (x @ W^T -> W rows are Bt[n][k])
// Tile BM=128 BN=64 BK=64; 256 thr = 8 waves (4x2), 2x2 frags, 8 wmma/tile/wave.
// Double-buffered LDS; tiles DMA'd by the Tensor Data Mover (TENSORcnt).
// =====================================================================
__global__ __launch_bounds__(256)
void proj_kernel(const __bf16* __restrict__ X, const __bf16* __restrict__ W,
                 const float* __restrict__ bias, const float* __restrict__ mask,
                 __bf16* __restrict__ Out, int apply_fmap, float post_scale) {
    constexpr int BM = 128, BN = 64, BK = 64;
    constexpr int KT = Cc / BK;                 // 16 k-tiles
    __shared__ __bf16 Asub[2][BM * BK];         // 2 x 16 KB
    __shared__ __bf16 Bsub[2][BN * BK];         // 2 x 8 KB

    const int tid = threadIdx.x;
    const int m0 = blockIdx.x * BM;
    const int n0 = blockIdx.y * BN;
    const int w  = tid >> 5;
    const int wr = w & 3;                       // m sub-block (x32)
    const int wc = w >> 2;                      // n sub-block (x32)

#if USE_TDM
    if (w == 0) {                               // wave 0 drives the TDM
        tdm_load_2d(lds_off_of(Asub[0]), X + (size_t)m0 * Cc + 0 * BK, BK, BM, Cc);
        tdm_load_2d(lds_off_of(Bsub[0]), W + (size_t)n0 * Cc + 0 * BK, BK, BN, Cc);
        tdm_load_2d(lds_off_of(Asub[1]), X + (size_t)m0 * Cc + 1 * BK, BK, BM, Cc);
        tdm_load_2d(lds_off_of(Bsub[1]), W + (size_t)n0 * Cc + 1 * BK, BK, BN, Cc);
    }
#endif

    v8f acc[2][2] = {};

    for (int kt = 0; kt < KT; ++kt) {
        const int buf = kt & 1;
#if USE_TDM
        if (w == 0) {                           // oldest 2 (this buf) must be done
            if (kt + 1 < KT) __builtin_amdgcn_s_wait_tensorcnt(2);   // next buf in flight
            else             __builtin_amdgcn_s_wait_tensorcnt(0);   // last tile: drain
        }
        __syncthreads();                        // publish buf to all waves
#else
        {   // fallback: synchronous bf16 b128 copy global -> LDS
            const int k0 = kt * BK;
            for (int i = tid; i < (BM * BK) / 8; i += 256) {
                const int r = i >> 3, c8 = i & 7;
                *(v4u*)&Asub[buf][r * BK + c8 * 8] =
                    *(const v4u*)&X[(size_t)(m0 + r) * Cc + k0 + c8 * 8];
            }
            for (int i = tid; i < (BN * BK) / 8; i += 256) {
                const int r = i >> 3, c8 = i & 7;
                *(v4u*)&Bsub[buf][r * BK + c8 * 8] =
                    *(const v4u*)&W[(size_t)(n0 + r) * Cc + k0 + c8 * 8];
            }
            __syncthreads();
        }
#endif
        const __bf16* Ab = Asub[buf];
        const __bf16* Bb = Bsub[buf];
#pragma unroll
        for (int ks = 0; ks < BK; ks += 32) {
            const v16bf a0 = load_frag_a(Ab + (wr * 32 + 0)  * BK + ks, BK);
            const v16bf a1 = load_frag_a(Ab + (wr * 32 + 16) * BK + ks, BK);
            const v16bf b0 = load_frag_b(Bb + (wc * 32 + 0)  * BK + ks, BK);
            const v16bf b1 = load_frag_b(Bb + (wc * 32 + 16) * BK + ks, BK);
            acc[0][0] = wmma_bf16(a0, b0, acc[0][0]);
            acc[0][1] = wmma_bf16(a0, b1, acc[0][1]);
            acc[1][0] = wmma_bf16(a1, b0, acc[1][0]);
            acc[1][1] = wmma_bf16(a1, b1, acc[1][1]);
        }
        __syncthreads();                        // all waves done reading buf
#if USE_TDM
        if (w == 0 && kt + 2 < KT) {            // refill buf for kt+2
            const int kn = kt + 2;
            tdm_load_2d(lds_off_of(Asub[buf]), X + (size_t)m0 * Cc + kn * BK, BK, BM, Cc);
            tdm_load_2d(lds_off_of(Bsub[buf]), W + (size_t)n0 * Cc + kn * BK, BK, BN, Cc);
        }
#endif
    }

    // epilogue: bias + feature map + mask + scale, store bf16
    const int lane  = tid & 31;
    const int col   = lane & 15;
    const int rbase = (lane >> 4) * 8;
    for (int mi = 0; mi < 2; ++mi) {
        for (int ni = 0; ni < 2; ++ni) {
            const int gn = n0 + wc * 32 + ni * 16 + col;
            const float bb = bias[gn];
            for (int j = 0; j < 8; ++j) {
                const int gm = m0 + wr * 32 + mi * 16 + rbase + j;
                float val = acc[mi][ni][j] + bb;
                if (apply_fmap) val = (val > 0.f) ? (val + 1.f) : __expf(val); // elu+1
                val *= mask[gm] * post_scale;
                Out[(size_t)gm * Cc + gn] = (__bf16)val;
            }
        }
    }
}

// =====================================================================
// Kernel 2: summed_k[b,h,d] = sum_l Kb[b,l,h,d]
// =====================================================================
__global__ __launch_bounds__(64)
void sumk_kernel(const __bf16* __restrict__ Kb, float* __restrict__ sumk) {
    const int bh = blockIdx.x;
    const int b = bh / Hc, h = bh % Hc;
    const int d = threadIdx.x;
    const __bf16* base = Kb + (size_t)b * Lc * Cc + h * Dc + d;
    float s = 0.f;
    for (int l = 0; l < Lc; ++l) s += (float)base[(size_t)l * Cc];
    sumk[bh * Dc + d] = s;
}

// =====================================================================
// Kernel 3: kv[b,h,d,e] = sum_l K[b,l,h,d] * V[b,l,h,e]   (K^T V, 64x64xL)
// =====================================================================
__global__ __launch_bounds__(512)
void kv_kernel(const __bf16* __restrict__ Kb, const __bf16* __restrict__ Vb,
               float* __restrict__ kvw) {
    constexpr int CH = 32;
    __shared__ __bf16 Kt[Dc * CH];         // Kt[d][l]
    __shared__ __bf16 Vt[Dc * CH];         // Vt[e][l]

    const int bh = blockIdx.x;
    const int b = bh / Hc, h = bh % Hc;
    const int tid = threadIdx.x;
    const int w  = tid >> 5;
    const int mi = w >> 2, ni = w & 3;

    const __bf16* Kbase = Kb + (size_t)b * Lc * Cc + h * Dc;
    const __bf16* Vbase = Vb + (size_t)b * Lc * Cc + h * Dc;

    v8f acc = {};
    for (int l0 = 0; l0 < Lc; l0 += CH) {
        for (int i = tid; i < Dc * CH; i += 512) {
            const int l = i >> 6, d = i & 63;
            const size_t g = (size_t)(l0 + l) * Cc + d;
            Kt[d * CH + l] = Kbase[g];
            Vt[d * CH + l] = Vbase[g];
        }
        __syncthreads();
        const v16bf a  = load_frag_a(&Kt[(mi * 16) * CH], CH);
        const v16bf bb = load_frag_b(&Vt[(ni * 16) * CH], CH);
        acc = wmma_bf16(a, bb, acc);
        __syncthreads();
    }

    const int lane = tid & 31;
    const int col = lane & 15, rbase = (lane >> 4) * 8;
    for (int j = 0; j < 8; ++j) {
        const int dd = mi * 16 + rbase + j;
        const int ee = ni * 16 + col;
        kvw[((size_t)bh * Dc + dd) * Dc + ee] = acc[j];
    }
}

// =====================================================================
// Kernel 4: denom[b,l,h] = 1 / (eps + q . summed_k)
// =====================================================================
__global__ __launch_bounds__(256)
void denom_kernel(const __bf16* __restrict__ Qb, const float* __restrict__ sumk,
                  float* __restrict__ denom) {
    const int n = blockIdx.x * 256 + threadIdx.x;
    if (n >= BLc * Hc) return;
    const int h = n % Hc;
    const int bl = n / Hc;
    const int b = bl / Lc;
    const __bf16* q = Qb + (size_t)bl * Cc + h * Dc;
    const float*  s = sumk + (b * Hc + h) * Dc;
    float a = 0.f;
    for (int d = 0; d < Dc; ++d) a += (float)q[d] * s[d];
    denom[n] = 1.0f / (EPSc + a);
}

// =====================================================================
// Kernel 5: out = (Q @ kv) * denom * L
// =====================================================================
__global__ __launch_bounds__(256)
void out_kernel(const __bf16* __restrict__ Qb, const float* __restrict__ kvw,
                const float* __restrict__ denomw, float* __restrict__ Out) {
    __shared__ __bf16 Qs[128 * Dc];
    __shared__ __bf16 kvT[Dc * Dc];

    const int blk = blockIdx.x;
    const int bh = blk >> 5;
    const int lt = blk & 31;
    const int b = bh / Hc, h = bh % Hc;
    const int m0 = lt * 128;
    const int tid = threadIdx.x;

    const float* kvp = kvw + (size_t)bh * Dc * Dc;
    for (int i = tid; i < Dc * Dc; i += 256) {
        const int d = i >> 6, e = i & 63;
        kvT[e * Dc + d] = (__bf16)kvp[i];
    }
    const __bf16* Qbase = Qb + (size_t)(b * Lc + m0) * Cc + h * Dc;
    for (int i = tid; i < 128 * 8; i += 256) {
        const int lr = i >> 3, c8 = i & 7;
        *(v4u*)&Qs[lr * Dc + c8 * 8] = *(const v4u*)&Qbase[(size_t)lr * Cc + c8 * 8];
    }
    __syncthreads();

    const int w = tid >> 5;
    v8f acc[4] = {};
#pragma unroll
    for (int k0 = 0; k0 < Dc; k0 += 32) {
        const v16bf a = load_frag_a(&Qs[(w * 16) * Dc + k0], Dc);
#pragma unroll
        for (int ni = 0; ni < 4; ++ni) {
            const v16bf bb = load_frag_b(&kvT[(ni * 16) * Dc + k0], Dc);
            acc[ni] = wmma_bf16(a, bb, acc[ni]);
        }
    }

    const int lane = tid & 31;
    const int col = lane & 15, rbase = (lane >> 4) * 8;
    for (int j = 0; j < 8; ++j) {
        const int l = m0 + w * 16 + rbase + j;
        const float scale = denomw[(size_t)(b * Lc + l) * Hc + h] * (float)Lc;
        float* orow = Out + ((size_t)(b * Lc + l) * Hc + h) * Dc;
        for (int ni = 0; ni < 4; ++ni)
            orow[ni * 16 + col] = acc[ni][j] * scale;
    }
}

// =====================================================================
extern "C" void kernel_launch(void* const* d_in, const int* in_sizes, int n_in,
                              void* d_out, int out_size, void* d_ws, size_t ws_size,
                              hipStream_t stream) {
    (void)in_sizes; (void)n_in; (void)out_size; (void)ws_size;

    const float* query  = (const float*)d_in[0];
    const float* key    = (const float*)d_in[1];
    const float* value  = (const float*)d_in[2];
    const float* mask_q = (const float*)d_in[3];
    const float* mask_kv= (const float*)d_in[4];
    const float* Wq = (const float*)d_in[5];
    const float* bq = (const float*)d_in[6];
    const float* Wk = (const float*)d_in[7];
    const float* bk = (const float*)d_in[8];
    const float* Wv = (const float*)d_in[9];
    const float* bv = (const float*)d_in[10];
    float* out = (float*)d_out;

    // workspace layout
    char* p = (char*)d_ws;
    __bf16* Qb  = (__bf16*)p; p += (size_t)BLc * Cc * sizeof(__bf16);
    __bf16* Kb  = (__bf16*)p; p += (size_t)BLc * Cc * sizeof(__bf16);
    __bf16* Vb  = (__bf16*)p; p += (size_t)BLc * Cc * sizeof(__bf16);
    __bf16* Xqb = (__bf16*)p; p += (size_t)BLc * Cc * sizeof(__bf16);
    __bf16* Xkb = (__bf16*)p; p += (size_t)BLc * Cc * sizeof(__bf16);
    __bf16* Xvb = (__bf16*)p; p += (size_t)BLc * Cc * sizeof(__bf16);
    __bf16* Wqb = (__bf16*)p; p += (size_t)Cc * Cc * sizeof(__bf16);
    __bf16* Wkb = (__bf16*)p; p += (size_t)Cc * Cc * sizeof(__bf16);
    __bf16* Wvb = (__bf16*)p; p += (size_t)Cc * Cc * sizeof(__bf16);
    float* kvw    = (float*)p; p += (size_t)Bc * Hc * Dc * Dc * sizeof(float);
    float* sumk   = (float*)p; p += (size_t)Bc * Hc * Dc * sizeof(float);
    float* denomw = (float*)p;

    // 0) one-shot fp32 -> bf16 conversions
    const int nX4 = (BLc * Cc) / 4, nW4 = (Cc * Cc) / 4;
    cvt_kernel<<<(nX4 + 255) / 256, 256, 0, stream>>>(query, Xqb, nX4);
    cvt_kernel<<<(nX4 + 255) / 256, 256, 0, stream>>>(key,   Xkb, nX4);
    cvt_kernel<<<(nX4 + 255) / 256, 256, 0, stream>>>(value, Xvb, nX4);
    cvt_kernel<<<(nW4 + 255) / 256, 256, 0, stream>>>(Wq, Wqb, nW4);
    cvt_kernel<<<(nW4 + 255) / 256, 256, 0, stream>>>(Wk, Wkb, nW4);
    cvt_kernel<<<(nW4 + 255) / 256, 256, 0, stream>>>(Wv, Wvb, nW4);

    // 1) projections — bf16 WMMA GEMMs, TDM-fed double-buffered LDS
    dim3 pg(BLc / 128, Cc / 64);
    proj_kernel<<<pg, 256, 0, stream>>>(Xqb, Wqb, bq, mask_q,  Qb, 1, 1.0f);
    proj_kernel<<<pg, 256, 0, stream>>>(Xkb, Wkb, bk, mask_kv, Kb, 1, 1.0f);
    proj_kernel<<<pg, 256, 0, stream>>>(Xvb, Wvb, bv, mask_kv, Vb, 0, 1.0f / (float)Lc);

    // 2) summed_k
    sumk_kernel<<<Bc * Hc, 64, 0, stream>>>(Kb, sumk);

    // 3) kv = K^T V  (WMMA)
    kv_kernel<<<Bc * Hc, 512, 0, stream>>>(Kb, Vb, kvw);

    // 4) denom
    denom_kernel<<<(BLc * Hc + 255) / 256, 256, 0, stream>>>(Qb, sumk, denomw);

    // 5) out = (Q @ kv) * denom * L  (WMMA)
    out_kernel<<<Bc * Hc * (Lc / 128), 256, 0, stream>>>(Qb, kvw, denomw, out);
}
